// SimpleReduNet_1898375544935
// MI455X (gfx1250) — matrix operand order repaired and checked by reading, not compiled, not run on hardware
//
#include <hip/hip_runtime.h>
#include <hip/hip_bf16.h>
#include <stdint.h>

// ---------------------------------------------------------------------------
// ReduNet forward on MI455X (gfx1250, wave32, WMMA bf16 -> f32 accumulate)
//   D=1024 (feature dim), N=4096 (samples), C=10 (classes), L=8 (layers)
// Per layer:
//   sims[c,n]  = ||Cs_c @ Z[:,n]||^2           (pass 1: 10 GEMMs, norm-reduce)
//   P[c,n]     = gamma_c * softmax_c(-sqrt(sims)/T)
//   G          = eta*(E@Z - sum_c (Cs_c@Z) diag(P[c,:]))   (pass 2: 11 GEMMs,
//                column scaling folded into the LDS B tile, one accumulator)
//   Z          = normalize_cols(Z + G)
// Tiling: 128x128 block, 256 thr = 8 wave32, wave patch 32x64 (2x4 WMMA),
//         TK=64 per LDS stage (2 k-chunks of 32), stride-36 u32 LDS padding
//         so all fragment reads are bank-conflict-free ds_load_b128.
// ---------------------------------------------------------------------------

#define RD_D 1024
#define RD_N 4096
#define RD_C 10
#define RD_L 8
#define LDS_STRIDE 36   // u32 per row: 32 k-pairs + 4 pad (mod4==0 -> 16B align)

typedef __attribute__((ext_vector_type(16))) __bf16 v16bf;
typedef __attribute__((ext_vector_type(8)))  float  v8f;

union BF16Frag { v16bf v; uint32_t u[8]; };

__device__ __forceinline__ uint32_t f2bf(float f) {
    uint32_t u = __float_as_uint(f);
    return (u + 0x7FFFu + ((u >> 16) & 1u)) >> 16;   // RNE bf16 truncation
}
__device__ __forceinline__ float bf2f(uint32_t b) {
    return __uint_as_float((b & 0xFFFFu) << 16);
}

// One K-stage (TK=64) of the 128x128 block tile:
//   A: f32 global -> bf16 LDS, row-major k-pairs, stride-36 padded
//   B: bf16 global -> LDS n-major [n][kpair] (+optional per-column scale)
//   then 2 k-chunks x 8 v_wmma_f32_16x16x32_bf16 per wave.
template <bool SCALE>
__device__ __forceinline__ void tile_kstage(const float* __restrict__ A,
                                            const unsigned short* __restrict__ Zbf,
                                            uint32_t* Alds, uint32_t* Blds,
                                            const float* wrow,
                                            int m0, int n0, int k0, int tid,
                                            v8f acc[2][4]) {
    __syncthreads();   // protect LDS from previous stage's readers
    // --- A tile: 128 rows x 64 k (f32 -> bf16). 2048 float4 / 256 threads.
#pragma unroll
    for (int it = 0; it < 8; ++it) {
        int idx = tid + it * 256;
        int m = idx >> 4, kq = idx & 15;
        const float* ap = A + (size_t)(m0 + m) * RD_D + k0 + kq * 4;
        const float4 f = *(const float4*)ap;
        if (k0 + 64 < RD_D)                    // warm L2 for next k-slab
            __builtin_prefetch(ap + 64, 0, 3); // -> global_prefetch_b8
        uint2 t;
        t.x = f2bf(f.x) | (f2bf(f.y) << 16);
        t.y = f2bf(f.z) | (f2bf(f.w) << 16);
        *(uint2*)(Alds + m * LDS_STRIDE + kq * 2) = t;
    }
    // --- B tile: 64 k x 128 n, stored n-major as u32 k-pairs [n][kpair].
#pragma unroll
    for (int it = 0; it < 8; ++it) {
        int idx = tid + it * 256;
        int kp = idx >> 6, np = idx & 63;
        int k = k0 + kp * 2, n = n0 + np * 2;
        uint32_t a = *(const uint32_t*)(Zbf + (size_t)k * RD_N + n);
        uint32_t b = *(const uint32_t*)(Zbf + (size_t)(k + 1) * RD_N + n);
        uint32_t p0, p1;  // p0: (k,k+1) at col n ; p1: at col n+1
        if (SCALE) {
            float w0 = wrow[np * 2 + 0], w1 = wrow[np * 2 + 1];
            p0 = f2bf(bf2f(a) * w0)       | (f2bf(bf2f(b) * w0) << 16);
            p1 = f2bf(bf2f(a >> 16) * w1) | (f2bf(bf2f(b >> 16) * w1) << 16);
        } else {
            p0 = (a & 0xFFFFu) | (b << 16);
            p1 = (a >> 16)     | (b & 0xFFFF0000u);
        }
        Blds[(np * 2 + 0) * LDS_STRIDE + kp] = p0;
        Blds[(np * 2 + 1) * LDS_STRIDE + kp] = p1;
    }
    __syncthreads();
    // --- fragments + MMA. Wave grid: 4 (M) x 2 (N); wave patch 32x64.
    int lane = tid & 31, wv = tid >> 5;
    int wm = wv >> 1, wn = wv & 1;
    int l15 = lane & 15, half = lane >> 4;
#pragma unroll
    for (int kh = 0; kh < 2; ++kh) {            // two 32-wide k-chunks
        BF16Frag af[2], bfr[4];
#pragma unroll
        for (int tm = 0; tm < 2; ++tm) {
            const uint32_t* ar = Alds + (wm * 32 + tm * 16 + l15) * LDS_STRIDE
                                + kh * 16 + half * 4;
#pragma unroll
            for (int v = 0; v < 8; ++v)          // ISA 16-bit A 16x32 layout:
                af[tm].u[v] = ar[(v >> 2) * 8 + (v & 3)]; // 2x contiguous quads
        }
#pragma unroll
        for (int tn = 0; tn < 4; ++tn) {
            const uint32_t* br = Blds + (wn * 64 + tn * 16 + l15) * LDS_STRIDE
                                + kh * 16 + half * 4;
#pragma unroll
            for (int v = 0; v < 8; ++v)          // B: same layout as A,
                bfr[tn].u[v] = br[(v >> 2) * 8 + (v & 3)]; // 2x contiguous quads
        }
#pragma unroll
        for (int tm = 0; tm < 2; ++tm)
#pragma unroll
            for (int tn = 0; tn < 4; ++tn)
                acc[tm][tn] = __builtin_amdgcn_wmma_f32_16x16x32_bf16(
                    false, af[tm].v, false, bfr[tn].v, (short)0,
                    acc[tm][tn], false, false);
    }
}

// --- Z(f32) -> bf16, and zero sims ------------------------------------------
__global__ __launch_bounds__(256) void k_cvt(const float* __restrict__ Zc,
                                             unsigned short* __restrict__ Zbf,
                                             float* __restrict__ sims) {
    int idx = blockIdx.x * 256 + threadIdx.x;
    Zbf[idx] = (unsigned short)f2bf(Zc[idx]);
    if (idx < RD_C * RD_N) sims[idx] = 0.0f;
}

// --- pass 1: sims[c,n] += column sum of squares of (Cs_c @ Z) tile ----------
__global__ __launch_bounds__(256) void k_sims(const float* __restrict__ CsL,
                                              const unsigned short* __restrict__ Zbf,
                                              float* __restrict__ sims) {
    __shared__ uint32_t Alds[128 * LDS_STRIDE];
    __shared__ uint32_t Blds[128 * LDS_STRIDE];
    __shared__ float simsp[128];
    int tid = threadIdx.x;
    int n0 = blockIdx.x * 128, m0 = blockIdx.y * 128, c = blockIdx.z;
    const float* A = CsL + (size_t)c * RD_D * RD_D;
    if (tid < 128) simsp[tid] = 0.0f;

    v8f acc[2][4] = {};
    for (int k0 = 0; k0 < RD_D; k0 += 64)
        tile_kstage<false>(A, Zbf, Alds, Blds, nullptr, m0, n0, k0, tid, acc);

    __syncthreads();
    int lane = tid & 31, wv = tid >> 5;
    int wn = wv & 1, l15 = lane & 15;
#pragma unroll
    for (int tn = 0; tn < 4; ++tn) {
        float s = 0.0f;
#pragma unroll
        for (int tm = 0; tm < 2; ++tm)
#pragma unroll
            for (int v = 0; v < 8; ++v) { float d = acc[tm][tn][v]; s += d * d; }
        atomicAdd(&simsp[wn * 64 + tn * 16 + l15], s);   // ds_add_f32
    }
    __syncthreads();
    if (tid < 128) atomicAdd(&sims[(size_t)c * RD_N + n0 + tid], simsp[tid]);
}

// --- softmax over classes, fold gamma ---------------------------------------
__global__ __launch_bounds__(256) void k_softmax(const float* __restrict__ sims,
                                                 const float* __restrict__ gamma,
                                                 const float* __restrict__ temp,
                                                 float* __restrict__ P) {
    int n = blockIdx.x * 256 + threadIdx.x;
    float invT = 1.0f / temp[0];
    float s[RD_C], m = -3.0e38f;
#pragma unroll
    for (int c = 0; c < RD_C; ++c) {
        s[c] = -sqrtf(fmaxf(sims[(size_t)c * RD_N + n], 0.0f)) * invT;
        m = fmaxf(m, s[c]);
    }
    float sum = 0.0f;
#pragma unroll
    for (int c = 0; c < RD_C; ++c) { s[c] = __expf(s[c] - m); sum += s[c]; }
    float inv = 1.0f / sum;
#pragma unroll
    for (int c = 0; c < RD_C; ++c) P[(size_t)c * RD_N + n] = gamma[c] * s[c] * inv;
}

// --- pass 2: G = eta*(E@Z) - eta*sum_c (Cs_c@Z) diag(gamma_c P_hat[c,:]) ----
__global__ __launch_bounds__(256) void k_term(const float* __restrict__ CsL,
                                              const float* __restrict__ EL,
                                              const unsigned short* __restrict__ Zbf,
                                              const float* __restrict__ P,
                                              const float* __restrict__ etaPtr,
                                              float* __restrict__ G) {
    __shared__ uint32_t Alds[128 * LDS_STRIDE];
    __shared__ uint32_t Blds[128 * LDS_STRIDE];
    __shared__ float wtab[(RD_C + 1) * 128];
    int tid = threadIdx.x;
    int n0 = blockIdx.x * 128, m0 = blockIdx.y * 128;
    float eta = etaPtr[0];
    for (int idx = tid; idx < (RD_C + 1) * 128; idx += 256) {
        int z = idx >> 7, n = idx & 127;
        wtab[idx] = (z < RD_C) ? (-eta * P[(size_t)z * RD_N + n0 + n]) : eta;
    }

    v8f acc[2][4] = {};
    for (int z = 0; z <= RD_C; ++z) {
        const float* A = (z < RD_C) ? (CsL + (size_t)z * RD_D * RD_D) : EL;
        const float* wrow = wtab + z * 128;
        for (int k0 = 0; k0 < RD_D; k0 += 64)
            tile_kstage<true>(A, Zbf, Alds, Blds, wrow, m0, n0, k0, tid, acc);
    }

    int lane = tid & 31, wv = tid >> 5;
    int wm = wv >> 1, wn = wv & 1;
    int l15 = lane & 15, half = lane >> 4;
#pragma unroll
    for (int tm = 0; tm < 2; ++tm)
#pragma unroll
        for (int tn = 0; tn < 4; ++tn) {
            int col = n0 + wn * 64 + tn * 16 + l15;
#pragma unroll
            for (int v = 0; v < 8; ++v) {
                int row = m0 + wm * 32 + tm * 16 + v + 8 * half;  // ISA C/D layout
                G[(size_t)row * RD_N + col] = acc[tm][tn][v];
            }
        }
}

// --- Z = normalize_cols(Z + G) ----------------------------------------------
__global__ __launch_bounds__(256) void k_update(float* __restrict__ Zc,
                                                const float* __restrict__ G) {
    int n = blockIdx.x * 256 + threadIdx.x;   // one thread per column (coalesced)
    float s = 0.0f;
    for (int i = 0; i < RD_D; ++i) {
        float v = Zc[(size_t)i * RD_N + n] + G[(size_t)i * RD_N + n];
        s += v * v;
    }
    float r = rsqrtf(s);
    for (int i = 0; i < RD_D; ++i) {
        size_t o = (size_t)i * RD_N + n;
        Zc[o] = (Zc[o] + G[o]) * r;
    }
}

extern "C" void kernel_launch(void* const* d_in, const int* in_sizes, int n_in,
                              void* d_out, int out_size, void* d_ws, size_t ws_size,
                              hipStream_t stream) {
    const float* Zin    = (const float*)d_in[0];   // (D,N)
    const float* Elist  = (const float*)d_in[1];   // (L,D,D)
    const float* Cslist = (const float*)d_in[2];   // (L,C,D,D)
    const float* gamma  = (const float*)d_in[3];   // (C,)
    const float* eta    = (const float*)d_in[4];   // scalar
    const float* temp   = (const float*)d_in[5];   // scalar
    // d_in[6] = L (==8, hardcoded; device int not readable during capture)
    float* out = (float*)d_out;

    const size_t DN = (size_t)RD_D * RD_N;
    char* w = (char*)d_ws;                                   // ws usage ~40.3 MB
    float*          Zcur = (float*)(w);                      // 16 MB
    float*          G    = (float*)(w + DN * 4);             // 16 MB
    unsigned short* Zbf  = (unsigned short*)(w + DN * 8);    //  8 MB
    float*          sims = (float*)(w + DN * 10);            // C*N f32
    float*          P    = sims + (size_t)RD_C * RD_N;       // C*N f32

    hipMemcpyAsync(Zcur, Zin, DN * sizeof(float), hipMemcpyDeviceToDevice, stream);

    for (int l = 0; l < RD_L; ++l) {
        const float* CsL = Cslist + (size_t)l * RD_C * RD_D * RD_D;
        const float* EL  = Elist  + (size_t)l * RD_D * RD_D;

        k_cvt<<<DN / 256, 256, 0, stream>>>(Zcur, Zbf, sims);
        k_sims<<<dim3(RD_N / 128, RD_D / 128, RD_C), 256, 0, stream>>>(CsL, Zbf, sims);
        k_softmax<<<RD_N / 256, 256, 0, stream>>>(sims, gamma, temp, P);
        k_term<<<dim3(RD_N / 128, RD_D / 128), 256, 0, stream>>>(CsL, EL, Zbf, P, eta, G);
        k_update<<<RD_N / 256, 256, 0, stream>>>(Zcur, G);
    }

    hipMemcpyAsync(out, Zcur, DN * sizeof(float), hipMemcpyDeviceToDevice, stream);
}
